// QuantSiglipMLP_35003983463157
// MI455X (gfx1250) — 8-bit matrix operand order repaired and checked by reading, compiled, not run on hardware
//
#include <hip/hip_runtime.h>
#include <stdint.h>

typedef int v8i __attribute__((ext_vector_type(8)));

#define N_TOK 23328
#define DH    1152
#define FI    4304
#define FP    4352
#define QMAXF 127.0f
#define TILES 8

__global__ __launch_bounds__(256) void quant_x_kernel(
    const float* __restrict__ x, int8_t* __restrict__ xq, float* __restrict__ sx) {
  const int row = blockIdx.x;
  const int tid = threadIdx.x;
  float v[5];
  float amax = 0.0f;
  int cnt = 0;
  for (int f = tid; f < DH; f += 256) {
    float val = x[(size_t)row * DH + f];
    v[cnt++] = val;
    amax = fmaxf(amax, fabsf(val));
  }
  __shared__ float red[256];
  red[tid] = amax;
  __syncthreads();
  for (int s = 128; s > 0; s >>= 1) {
    if (tid < s) red[tid] = fmaxf(red[tid], red[tid + s]);
    __syncthreads();
  }
  const float scale = fmaxf(red[0] / QMAXF, 1e-8f);
  if (tid == 0) sx[row] = scale;
  const float inv = 1.0f / scale;
  cnt = 0;
  for (int f = tid; f < DH; f += 256) {
    float q = fminf(fmaxf(rintf(v[cnt++] * inv), -QMAXF), QMAXF);
    xq[(size_t)row * DH + f] = (int8_t)q;
  }
}

__global__ void pack_w1_kernel(const int* __restrict__ w1, int8_t* __restrict__ w1p) {
  size_t i = (size_t)blockIdx.x * 256 + threadIdx.x;
  if (i >= (size_t)FP * DH) return;
  int f = (int)(i / DH);
  int d = (int)(i % DH);
  w1p[i] = (f < FI) ? (int8_t)w1[(size_t)f * DH + d] : (int8_t)0;
}

__global__ void pack_w2_kernel(const int* __restrict__ w2, int8_t* __restrict__ w2p) {
  size_t i = (size_t)blockIdx.x * 256 + threadIdx.x;
  if (i >= (size_t)DH * FP) return;
  int d = (int)(i / FP);
  int f = (int)(i % FP);
  w2p[i] = (f < FI) ? (int8_t)w2[(size_t)d * FI + f] : (int8_t)0;
}

__device__ __forceinline__ v8i load_a_frag(const int8_t* base) {
  const int2* ap = (const int2*)base;
  int2 a0 = ap[0];
  int2 a1 = ap[1];
  int2 a2 = ap[4];
  int2 a3 = ap[5];
  v8i A;
  A[0] = a0.x; A[1] = a0.y;
  A[2] = a1.x; A[3] = a1.y;
  A[4] = a2.x; A[5] = a2.y;
  A[6] = a3.x; A[7] = a3.y;
  return A;
}

__device__ __forceinline__ v8i load_b_frag(const int8_t* base) {
  const int4* bp = (const int4*)base;
  int4 lo = bp[0];
  int4 hi = bp[2];
  v8i B;
  B[0] = lo.x; B[1] = lo.y; B[2] = lo.z; B[3] = lo.w;
  B[4] = hi.x; B[5] = hi.y; B[6] = hi.z; B[7] = hi.w;
  return B;
}

__global__ __launch_bounds__(256) void gemm1_kernel(
    const int8_t* __restrict__ xq, const float* __restrict__ sx,
    const int8_t* __restrict__ w1p, const float* __restrict__ s_w1,
    const float* __restrict__ b1, float* __restrict__ act) {
  const int lane = threadIdx.x & 31;
  const int wave = threadIdx.x >> 5;
  const int rowTile = blockIdx.y * 8 + wave;
  if (rowTile >= N_TOK / 16) return;

  const int m0 = rowTile * 16;
  const int f0 = blockIdx.x * (16 * TILES);
  const int hi = lane >> 4;
  const int lc = lane & 15;

  v8i acc[TILES];
#pragma unroll
  for (int t = 0; t < TILES; ++t) acc[t] = (v8i){0, 0, 0, 0, 0, 0, 0, 0};

  const int8_t* aptr = xq + (size_t)(m0 + lc) * DH + hi * 16;
  const int8_t* bptr[TILES];
#pragma unroll
  for (int t = 0; t < TILES; ++t)
    bptr[t] = w1p + (size_t)(f0 + t * 16 + lc) * DH + hi * 16;

  v8i A = load_a_frag(aptr);
  v8i B[TILES];
#pragma unroll
  for (int t = 0; t < TILES; ++t) B[t] = load_b_frag(bptr[t]);

#pragma unroll 2
  for (int k0 = 0; k0 < DH; k0 += 64) {
    v8i An = load_a_frag(aptr + k0 + 64);
    v8i Bn[TILES];
#pragma unroll
    for (int t = 0; t < TILES; ++t) Bn[t] = load_b_frag(bptr[t] + k0 + 64);
#pragma unroll
    for (int t = 0; t < TILES; ++t)
      acc[t] = __builtin_amdgcn_wmma_i32_16x16x64_iu8(
          true, A, true, B[t], acc[t], false, false);
    A = An;
#pragma unroll
    for (int t = 0; t < TILES; ++t) B[t] = Bn[t];
  }

  float sxv[8];
#pragma unroll
  for (int g = 0; g < 8; ++g) sxv[g] = sx[m0 + g + hi * 8];

#pragma unroll
  for (int t = 0; t < TILES; ++t) {
    const int fb = f0 + t * 16;
    if (fb >= FI) continue;
    const int col = fb + lc;
    const float sw = s_w1[col];
    const float bb = b1[col];
#pragma unroll
    for (int g = 0; g < 8; ++g) {
      const int row = m0 + g + hi * 8;
      float h = (float)acc[t][g] * (sxv[g] * sw) + bb;
      float u = 0.7978845608028654f * (h + 0.044715f * h * h * h);
      float a = 0.5f * h * (1.0f + tanhf(u));
      act[(size_t)row * FP + col] = a;
    }
  }
}

__global__ __launch_bounds__(256) void quant_a_kernel(
    const float* __restrict__ act, int8_t* __restrict__ aq, float* __restrict__ sa) {
  const int row = blockIdx.x;
  const int tid = threadIdx.x;
  float v[17];
  float amax = 0.0f;
  int cnt = 0;
  for (int f = tid; f < FI; f += 256) {
    float val = act[(size_t)row * FP + f];
    v[cnt++] = val;
    amax = fmaxf(amax, fabsf(val));
  }
  __shared__ float red[256];
  red[tid] = amax;
  __syncthreads();
  for (int s = 128; s > 0; s >>= 1) {
    if (tid < s) red[tid] = fmaxf(red[tid], red[tid + s]);
    __syncthreads();
  }
  const float scale = fmaxf(red[0] / QMAXF, 1e-8f);
  if (tid == 0) sa[row] = scale;
  const float inv = 1.0f / scale;
  cnt = 0;
  for (int f = tid; f < FI; f += 256) {
    float q = fminf(fmaxf(rintf(v[cnt++] * inv), -QMAXF), QMAXF);
    aq[(size_t)row * FP + f] = (int8_t)q;
  }
  if (tid < FP - FI) aq[(size_t)row * FP + FI + tid] = 0;
}

__global__ __launch_bounds__(256) void gemm2_kernel(
    const int8_t* __restrict__ aq, const float* __restrict__ sa,
    const int8_t* __restrict__ w2p, const float* __restrict__ s_w2,
    const float* __restrict__ b2, float* __restrict__ out) {
  const int lane = threadIdx.x & 31;
  const int wave = threadIdx.x >> 5;
  const int rowTile = blockIdx.y * 8 + wave;
  if (rowTile >= N_TOK / 16) return;

  const int m0 = rowTile * 16;
  const int d0 = blockIdx.x * (16 * TILES);
  const int hi = lane >> 4;
  const int lc = lane & 15;

  v8i acc[TILES];
#pragma unroll
  for (int t = 0; t < TILES; ++t) acc[t] = (v8i){0, 0, 0, 0, 0, 0, 0, 0};

  const int8_t* aptr = aq + (size_t)(m0 + lc) * FP + hi * 16;
  const int8_t* bptr[TILES];
#pragma unroll
  for (int t = 0; t < TILES; ++t)
    bptr[t] = w2p + (size_t)(d0 + t * 16 + lc) * FP + hi * 16;

  v8i A = load_a_frag(aptr);
  v8i B[TILES];
#pragma unroll
  for (int t = 0; t < TILES; ++t) B[t] = load_b_frag(bptr[t]);

#pragma unroll 2
  for (int k0 = 0; k0 < FP; k0 += 64) {
    v8i An = load_a_frag(aptr + k0 + 64);
    v8i Bn[TILES];
#pragma unroll
    for (int t = 0; t < TILES; ++t) Bn[t] = load_b_frag(bptr[t] + k0 + 64);
#pragma unroll
    for (int t = 0; t < TILES; ++t)
      acc[t] = __builtin_amdgcn_wmma_i32_16x16x64_iu8(
          true, A, true, B[t], acc[t], false, false);
    A = An;
#pragma unroll
    for (int t = 0; t < TILES; ++t) B[t] = Bn[t];
  }

  float sav[8];
#pragma unroll
  for (int g = 0; g < 8; ++g) sav[g] = sa[m0 + g + hi * 8];

#pragma unroll
  for (int t = 0; t < TILES; ++t) {
    const int col = d0 + t * 16 + lc;
    const float sw = s_w2[col];
    const float bb = b2[col];
#pragma unroll
    for (int g = 0; g < 8; ++g) {
      const int row = m0 + g + hi * 8;
      out[(size_t)row * DH + col] = (float)acc[t][g] * (sav[g] * sw) + bb;
    }
  }
}

extern "C" void kernel_launch(void* const* d_in, const int* in_sizes, int n_in,
                              void* d_out, int out_size, void* d_ws, size_t ws_size,
                              hipStream_t stream) {
  const float* x    = (const float*)d_in[0];
  const int*   w1q  = (const int*)  d_in[1];
  const float* s_w1 = (const float*)d_in[2];
  const float* b1   = (const float*)d_in[3];
  const int*   w2q  = (const int*)  d_in[4];
  const float* s_w2 = (const float*)d_in[5];
  const float* b2   = (const float*)d_in[6];
  float* out = (float*)d_out;

  char* ws = (char*)d_ws;
  size_t off = 0;
  auto carve = [&](size_t bytes) -> char* {
    char* p = ws + off;
    off += (bytes + 256 + 255) & ~(size_t)255;
    return p;
  };
  int8_t* xq  = (int8_t*)carve((size_t)N_TOK * DH);
  float*  sx  = (float*) carve((size_t)N_TOK * 4);
  int8_t* w1p = (int8_t*)carve((size_t)FP * DH);
  int8_t* w2p = (int8_t*)carve((size_t)DH * FP);
  float*  act = (float*) carve((size_t)N_TOK * FP * 4);
  int8_t* aq  = (int8_t*)carve((size_t)N_TOK * FP);
  float*  sa  = (float*) carve((size_t)N_TOK * 4);

  quant_x_kernel<<<N_TOK, 256, 0, stream>>>(x, xq, sx);

  {
    int blks = (int)(((size_t)FP * DH + 255) / 256);
    pack_w1_kernel<<<blks, 256, 0, stream>>>(w1q, w1p);
  }
  {
    int blks = (int)(((size_t)DH * FP + 255) / 256);
    pack_w2_kernel<<<blks, 256, 0, stream>>>(w2q, w2p);
  }

  gemm1_kernel<<<dim3(FP / (16 * TILES), (N_TOK / 16 + 7) / 8), 256, 0, stream>>>(
      xq, sx, w1p, s_w1, b1, act);

  quant_a_kernel<<<N_TOK, 256, 0, stream>>>(act, aq, sa);

  gemm2_kernel<<<dim3(DH / (16 * TILES), (N_TOK / 16 + 7) / 8), 256, 0, stream>>>(
      aq, sa, w2p, s_w2, b2, out);
}
